// LossFunction_55765855371815
// MI455X (gfx1250) — compile-verified, matrix operand order checked by reading, not run on hardware
//
#include <hip/hip_runtime.h>

typedef float      v4f  __attribute__((ext_vector_type(4)));
typedef __attribute__((ext_vector_type(8)))  float    v8f;
typedef __attribute__((ext_vector_type(16))) _Float16 v16h;

// ---------------------------------------------------------------------------
// Kernel 1: streaming pass. One wave32 per row (V=128 -> one float4 per lane).
// Per block (8 waves x 16 rows = 128 rows): emit 3 fp32 partials
//   partials[b]              = sum of masked NLL
//   partials[nblocks + b]    = count of non-pad targets
//   partials[2*nblocks + b]  = sum of cluster penalties
// ---------------------------------------------------------------------------
__global__ __launch_bounds__(256) void ce_rows_kernel(
    const float* __restrict__ preds, const int* __restrict__ targets,
    const int* __restrict__ ctab, float* __restrict__ partials,
    int N, int nblocks)
{
    __shared__ int   s_ctab[128];
    __shared__ float s_part[3][8];

    const int tid  = threadIdx.x;
    const int wave = tid >> 5;
    const int lane = tid & 31;

    if (tid < 128) s_ctab[tid] = ctab[tid];
    __syncthreads();

    const int rowBase = blockIdx.x * 128 + wave * 16;

    // Preload this wave's 16 targets once (lanes 0..15), broadcast via shfl.
    int t_pre = 0;
    if (lane < 16 && (rowBase + lane) < N) t_pre = targets[rowBase + lane];

    float w_nll = 0.0f, w_mask = 0.0f, w_pen = 0.0f;

    for (int r = 0; r < 16; ++r) {
        const int row = rowBase + r;
        if (row >= N) break;

        // 512B row, read-once stream: non-temporal B128 load, fully coalesced.
        const v4f* p = reinterpret_cast<const v4f*>(preds) + (size_t)row * 32;
        v4f v = __builtin_nontemporal_load(p + lane);

        // Lane-local max / first-occurrence argmax over 4 values.
        float m  = v.x; int mi = lane * 4;
        if (v.y > m) { m = v.y; mi = lane * 4 + 1; }
        if (v.z > m) { m = v.z; mi = lane * 4 + 2; }
        if (v.w > m) { m = v.w; mi = lane * 4 + 3; }

        // wave32 butterfly: global max, lowest index on ties (argmax semantics).
        #pragma unroll
        for (int off = 16; off >= 1; off >>= 1) {
            float om = __shfl_xor(m,  off, 32);
            int   oi = __shfl_xor(mi, off, 32);
            if (om > m || (om == m && oi < mi)) { m = om; mi = oi; }
        }

        float e = __expf(v.x - m) + __expf(v.y - m) +
                  __expf(v.z - m) + __expf(v.w - m);

        const int t  = __shfl(t_pre, r, 32);
        const int lo = lane * 4;
        float xt = 0.0f;
        if (t >= lo && t < lo + 4)
            xt = (t == lo) ? v.x : (t == lo + 1) ? v.y : (t == lo + 2) ? v.z : v.w;

        #pragma unroll
        for (int off = 16; off >= 1; off >>= 1) {
            e  += __shfl_xor(e,  off, 32);
            xt += __shfl_xor(xt, off, 32);
        }

        // -log_softmax[t] = max + log(sum exp) - x_t   (uniform across wave)
        const float nll = (m + __logf(e)) - xt;
        if (t != 0) { w_nll += nll; w_mask += 1.0f; }
        w_pen += (mi == t) ? 0.0f : ((s_ctab[mi] == s_ctab[t]) ? 0.5f : 1.0f);
    }

    if (lane == 0) {
        s_part[0][wave] = w_nll; s_part[1][wave] = w_mask; s_part[2][wave] = w_pen;
    }
    __syncthreads();
    if (tid == 0) {
        float a = 0.f, b = 0.f, c = 0.f;
        #pragma unroll
        for (int w = 0; w < 8; ++w) { a += s_part[0][w]; b += s_part[1][w]; c += s_part[2][w]; }
        partials[blockIdx.x]               = a;
        partials[nblocks + blockIdx.x]     = b;
        partials[2 * nblocks + blockIdx.x] = c;
    }
}

// ---------------------------------------------------------------------------
// Kernel 2: single wave32.
//  - NLL partials: deterministic fp32 lane-strided + butterfly reduction.
//  - mask/penalty partials: v_wmma_f32_16x16x32_f16 with B = ones.
//    Block partials are integers (<=128) / multiples of 0.5 (<=128): EXACT in
//    f16; C accumulates in f32 (totals < 2^19 half-steps < 2^24): EXACT.
//    B all-ones => every A element hits every D column once, so A packing
//    order is irrelevant; each lane packs 16 disjoint values.
// ---------------------------------------------------------------------------
__global__ __launch_bounds__(32) void finalize_kernel(
    const float* __restrict__ partials, float* __restrict__ out,
    int nblocks, float invN)
{
    const int lane = threadIdx.x;

    // ---- fp32 NLL sum (deterministic fixed order) ----
    float s = 0.0f;
    for (int i = lane; i < nblocks; i += 32) s += partials[i];
    #pragma unroll
    for (int off = 16; off >= 1; off >>= 1) s += __shfl_xor(s, off, 32);

    // ---- WMMA exact sums for mask count and penalty ----
    const float* pm = partials + nblocks;
    const float* pp = partials + 2 * nblocks;

    v16h ones;
    #pragma unroll
    for (int j = 0; j < 16; ++j) ones[j] = (_Float16)1.0f;

    v8f cm = {}; v8f cp = {};
    const int chunks = nblocks >> 9;              // 512 values per WMMA step
    for (int it = 0; it < chunks; ++it) {
        const int base = (it << 9) + lane * 16;
        v16h am, ap;
        #pragma unroll
        for (int j = 0; j < 16; ++j) {
            am[j] = (_Float16)pm[base + j];
            ap[j] = (_Float16)pp[base + j];
        }
        // D = A x ones + C : row-sums broadcast to every column.
        cm = __builtin_amdgcn_wmma_f32_16x16x32_f16(false, am, false, ones,
                                                    (short)0, cm, false, false);
        cp = __builtin_amdgcn_wmma_f32_16x16x32_f16(false, ap, false, ones,
                                                    (short)0, cp, false, false);
    }
    // Column n of D lives in lane n (M=0..7 in v0..7) + lane n+16 (M=8..15).
    float sm = cm[0]+cm[1]+cm[2]+cm[3]+cm[4]+cm[5]+cm[6]+cm[7];
    float sp = cp[0]+cp[1]+cp[2]+cp[3]+cp[4]+cp[5]+cp[6]+cp[7];
    sm += __shfl_xor(sm, 16, 32);
    sp += __shfl_xor(sp, 16, 32);

    // Scalar tail (nblocks not a multiple of 512) — deterministic.
    float tm = 0.0f, tp = 0.0f;
    for (int i = (chunks << 9) + lane; i < nblocks; i += 32) { tm += pm[i]; tp += pp[i]; }
    #pragma unroll
    for (int off = 16; off >= 1; off >>= 1) {
        tm += __shfl_xor(tm, off, 32);
        tp += __shfl_xor(tp, off, 32);
    }
    sm += tm; sp += tp;

    if (lane == 0) {
        const float ce = s / fmaxf(sm, 1.0f);
        const float cc = sp * invN;
        const float om = 1.0f - __expf(-cc);
        out[0] = ce + om * om * cc;     // focal scaling, GAMMA = 2
    }
}

// ---------------------------------------------------------------------------
extern "C" void kernel_launch(void* const* d_in, const int* in_sizes, int n_in,
                              void* d_out, int out_size, void* d_ws, size_t ws_size,
                              hipStream_t stream) {
    const float* preds   = (const float*)d_in[0];
    const int*   targets = (const int*)  d_in[1];
    const int*   ctab    = (const int*)  d_in[2];
    float*       out     = (float*)d_out;
    float*       ws      = (float*)d_ws;       // 3 * nblocks fp32 partials

    const int N       = in_sizes[1];           // 262144 rows
    const int nblocks = (N + 127) / 128;       // 128 rows per block -> 2048

    ce_rows_kernel<<<nblocks, 256, 0, stream>>>(preds, targets, ctab, ws, N, nblocks);
    finalize_kernel<<<1, 32, 0, stream>>>(ws, out, nblocks, 1.0f / (float)N);
}
